// HyenaOperator_18494129177060
// MI455X (gfx1250) — matrix operand order repaired and checked by reading, not compiled, not run on hardware
//
#include <hip/hip_runtime.h>
#include <hip/hip_bf16.h>
#include <math.h>

// ---------------------------------------------------------------------------
// HyenaOperator for MI455X (gfx1250, wave32)
//   B=8, C=2, H=1024, L=2048, n=2L=4096
//   FFT long-conv (bandwidth bound, ~0.6 GB -> ~27us @ 23.3 TB/s)
//   + 16384x2048x2048 projection GEMM on v_wmma_f32_16x16x32_bf16
//   Fragments staged in K-contiguous LDS -> ds_load_b128 (no scalar gathers)
// ---------------------------------------------------------------------------

#define LAM   0.1f
#define Bb    8
#define Cc    2
#define Hh    1024
#define Ll    2048
#define Nfft  4096
#define NBIN  2049          // rfft bins (n/2+1)
#define CH    (Cc*Hh)       // 2048 (GEMM K)
#define OUTD  (2*Hh)        // 2048 (GEMM N)
#define NT    512           // threads for FFT kernels

typedef __attribute__((ext_vector_type(16))) __bf16          v16bf;
typedef __attribute__((ext_vector_type(16))) unsigned short  v16us;
typedef __attribute__((ext_vector_type(8)))  float           v8f;

union FragQ {                 // 32B fragment: two 16B LDS loads -> WMMA operand
  uint4  q[2];
  v16us  us;
  v16bf  bf;
};

__device__ __forceinline__ unsigned short f2bf(float f) {
  unsigned int x = __float_as_uint(f);
  x += 0x7FFFu + ((x >> 16) & 1u);         // round-to-nearest-even
  return (unsigned short)(x >> 16);
}

// ---------------------------------------------------------------------------
// In-LDS 4096-point complex Stockham FFT (self-sorting, ping-pong buffers).
// dsign = -1 forward, +1 inverse (unnormalized). 12 stages (even) -> result
// ends back in (sre,sim). Must be run with NT threads.
// ---------------------------------------------------------------------------
__device__ void fft4096(float* sre, float* sim, float* tre, float* tim, float dsign) {
  float* ar = sre; float* ai = sim; float* br = tre; float* bi = tim;
  const float w0 = dsign * 1.5339807878856412e-3f;   // 2*pi/4096
  #pragma unroll 1
  for (int s = 0; s < 12; ++s) {
    const int m = 1 << s;
    __syncthreads();
    #pragma unroll
    for (int r = 0; r < 2048 / NT; ++r) {
      const int idx = threadIdx.x + r * NT;
      const int jm = idx & ~(m - 1);                 // j*m
      float x0r = ar[idx],        x0i = ai[idx];
      float x1r = ar[idx + 2048], x1i = ai[idx + 2048];
      float sr = x0r + x1r, si = x0i + x1i;
      float dr = x0r - x1r, di = x0i - x1i;
      float cw, sw; __sincosf(w0 * (float)jm, &sw, &cw);
      const int o0 = idx + jm;
      br[o0]     = sr;              bi[o0]     = si;
      br[o0 + m] = cw * dr - sw * di;
      bi[o0 + m] = sw * dr + cw * di;
    }
    float* t;
    t = ar; ar = br; br = t;
    t = ai; ai = bi; bi = t;
  }
  __syncthreads();
}

// ---------------------------------------------------------------------------
// Kernel 1: soft-threshold kernel rows + forward FFT -> k_f  (C*H blocks)
// ---------------------------------------------------------------------------
__global__ __launch_bounds__(NT)
void k_prep_fft(const float* __restrict__ kern, float* __restrict__ kf) {
  __shared__ float sre[4096], sim[4096], tre[4096], tim[4096];
  const int row = blockIdx.x;                        // c*H + h
  const float* kr = kern + (size_t)row * Nfft;
  #pragma unroll
  for (int r = 0; r < Nfft / NT; ++r) {
    const int i = threadIdx.x + r * NT;
    float v = kr[i];
    v = copysignf(fmaxf(fabsf(v) - LAM, 0.0f), v);   // shrinkage
    sre[i] = v; sim[i] = 0.0f;
  }
  fft4096(sre, sim, tre, tim, -1.0f);
  float* dst = kf + (size_t)row * (2 * NBIN);
  for (int f = threadIdx.x; f < NBIN; f += NT) {
    dst[2 * f] = sre[f]; dst[2 * f + 1] = sim[f];
  }
}

// ---------------------------------------------------------------------------
// Kernel 2: zero-pad u rows + forward FFT -> u_f  (B*H blocks)
// ---------------------------------------------------------------------------
__global__ __launch_bounds__(NT)
void k_ufft(const float* __restrict__ u, float* __restrict__ uf) {
  __shared__ float sre[4096], sim[4096], tre[4096], tim[4096];
  const int row = blockIdx.x;                        // b*H + h
  const float* ur = u + (size_t)row * Ll;
  #pragma unroll
  for (int r = 0; r < Nfft / NT; ++r) {
    const int i = threadIdx.x + r * NT;
    sre[i] = (i < Ll) ? ur[i] : 0.0f;
    sim[i] = 0.0f;
  }
  fft4096(sre, sim, tre, tim, -1.0f);
  float* dst = uf + (size_t)row * (2 * NBIN);
  for (int f = threadIdx.x; f < NBIN; f += NT) {
    dst[2 * f] = sre[f]; dst[2 * f + 1] = sim[f];
  }
}

// ---------------------------------------------------------------------------
// Kernel 3: spectral product + Hermitian mirror + inverse FFT + D bypass +
//           erf-GELU -> bf16 activations act[b, k=c*H+h, l] (B*C*H blocks)
// ---------------------------------------------------------------------------
__global__ __launch_bounds__(NT)
void k_conv(const float* __restrict__ uf, const float* __restrict__ kf,
            const float* __restrict__ u,  const float* __restrict__ Dp,
            unsigned short* __restrict__ act) {
  __shared__ float sre[4096], sim[4096], tre[4096], tim[4096];
  const int gid = blockIdx.x;                        // b*C*H + c*H + h
  const int b = gid >> 11;
  const int c = (gid >> 10) & 1;
  const int h = gid & 1023;
  const float* ufr = uf + (size_t)(b * Hh + h) * (2 * NBIN);
  const float* kfr = kf + (size_t)(c * Hh + h) * (2 * NBIN);
  for (int f = threadIdx.x; f < NBIN; f += NT) {
    float ur = ufr[2 * f], ui = ufr[2 * f + 1];
    float kr = kfr[2 * f], ki = kfr[2 * f + 1];
    float yr = ur * kr - ui * ki;
    float yi = ur * ki + ui * kr;
    sre[f] = yr; sim[f] = yi;
    if (f >= 1 && f <= 2047) { sre[4096 - f] = yr; sim[4096 - f] = -yi; }
  }
  fft4096(sre, sim, tre, tim, +1.0f);                // unnormalized inverse
  const float dv = Dp[c * Hh + h];
  const float* ur = u + (size_t)(b * Hh + h) * Ll;
  unsigned short* dst = act + ((size_t)b * CH + (c * Hh + h)) * Ll;
  #pragma unroll
  for (int r = 0; r < Ll / NT; ++r) {
    const int l = threadIdx.x + r * NT;
    float y = sre[l] * (1.0f / 4096.0f) + dv * ur[l];
    float g = 0.5f * y * (1.0f + erff(y * 0.70710678118654752f));  // exact GELU
    dst[l] = f2bf(g);
  }
}

// ---------------------------------------------------------------------------
// Kernel 4: projection GEMM  z[b,l,o] = sum_k act[b,k,l] * W[o,k] + bias[o]
//   M=2048/b (l), N=2048 (o), K=2048. Block tile 128x64, 8 waves, each wave
//   a 2x2 grid of 16x16 tiles; K stepped by 32 via v_wmma_f32_16x16x32_bf16.
//   LDS is K-contiguous so every fragment is two ds_load_b128:
//     A (16x32, 16-bit): lane m=lane&15, K at kbase=8*(lane>=16): [kb..kb+7],
//                        [kb+16..kb+23]  (ISA A-striping table)
//     B (32x16, 16-bit): lane n=lane&15, K at kbase=8*(lane>=16): [kb..kb+7], [kb+16..kb+23]
//                        (same ISA table order as A)
// ---------------------------------------------------------------------------
#define TM  128
#define TN  64
#define TK  32
#define LDK 40    // K row stride (ushorts): 80B, 16B-aligned, bank-skewed

__global__ __launch_bounds__(256)
void k_gemm(const unsigned short* __restrict__ act, const float* __restrict__ W,
            const float* __restrict__ bias, float* __restrict__ z) {
  __shared__ unsigned short sA[TM][LDK];   // [m][k]  128x40 = 10.0 KB
  __shared__ unsigned short sB[TN][LDK];   // [n][k]   64x40 =  5.0 KB
  const int m0   = blockIdx.x * TM;
  const int n0   = blockIdx.y * TN;
  const int b    = blockIdx.z;
  const int wave = threadIdx.x >> 5;
  const int lane = threadIdx.x & 31;
  const int wm   = wave & 3;               // 0..3 -> M offset wm*32
  const int wn   = wave >> 2;              // 0..1 -> N offset wn*32
  const int r16  = lane & 15;
  const int hi   = lane >> 4;              // half-wave select
  const int kbA  = hi * 8;                 // A frag K base
  const int kbB  = hi * 8;                 // B frag K base

  const unsigned short* actB = act + (size_t)b * CH * Ll;   // + k*Ll + l

  // B staging decomposition: thread -> (n row, 8-wide k chunk)
  const int bn = threadIdx.x >> 2;         // 0..63
  const int bk = (threadIdx.x & 3) << 3;   // 0,8,16,24

  v8f acc[2][2] = {};

  for (int k0 = 0; k0 < CH; k0 += TK) {
    // ---- A tile: 32 K-rows x 128 l-cols; coalesced 16B global loads,
    //      transposed into sA[m][k] (b16 LDS stores, off the critical path)
    #pragma unroll
    for (int rep = 0; rep < 2; ++rep) {
      const int i  = threadIdx.x + rep * 256;   // 0..511
      const int kk = i >> 4;                    // 0..31
      const int mv = (i & 15) << 3;             // 0,8,...,120
      uint4 pk = *(const uint4*)(actB + (size_t)(k0 + kk) * Ll + m0 + mv);
      const unsigned short* pv = (const unsigned short*)&pk;
      #pragma unroll
      for (int e = 0; e < 8; ++e) sA[mv + e][kk] = pv[e];
    }
    // ---- B tile: W[o,k] f32 -> bf16 into sB[n][k]; float4 loads + one
    //      packed ds_store_b128 per thread
    {
      const float* wp = &W[(size_t)(n0 + bn) * CH + k0 + bk];
      float4 w0 = *(const float4*)(wp);
      float4 w1 = *(const float4*)(wp + 4);
      union { unsigned short s[8]; uint4 q; } pk;
      pk.s[0] = f2bf(w0.x); pk.s[1] = f2bf(w0.y);
      pk.s[2] = f2bf(w0.z); pk.s[3] = f2bf(w0.w);
      pk.s[4] = f2bf(w1.x); pk.s[5] = f2bf(w1.y);
      pk.s[6] = f2bf(w1.z); pk.s[7] = f2bf(w1.w);
      *(uint4*)&sB[bn][bk] = pk.q;
    }
    if (k0 + TK < CH)
      __builtin_prefetch(actB + (size_t)(k0 + TK) * Ll + m0, 0, 1); // global_prefetch_b8
    __syncthreads();

    // ---- fragments: two 16B LDS vector loads each (rows are 16B-aligned)
    FragQ af[2], bg[2];
    #pragma unroll
    for (int t = 0; t < 2; ++t) {
      const int m = wm * 32 + t * 16 + r16;
      const int n = wn * 32 + t * 16 + r16;
      af[t].q[0] = *(const uint4*)&sA[m][kbA];
      af[t].q[1] = *(const uint4*)&sA[m][kbA + 16];
      bg[t].q[0] = *(const uint4*)&sB[n][kbB];
      bg[t].q[1] = *(const uint4*)&sB[n][kbB + 16];
    }
    #pragma unroll
    for (int tm = 0; tm < 2; ++tm)
      #pragma unroll
      for (int tn = 0; tn < 2; ++tn)
        acc[tm][tn] = __builtin_amdgcn_wmma_f32_16x16x32_bf16(
            false, af[tm].bf, false, bg[tn].bf,
            (short)0, acc[tm][tn], false, false);
    __syncthreads();
  }

  // ---- epilogue: C/D layout (VGPR r -> M=r+8*(lane>=16), N=lane&15)
  #pragma unroll
  for (int tm = 0; tm < 2; ++tm) {
    #pragma unroll
    for (int tn = 0; tn < 2; ++tn) {
      const int ng = n0 + wn * 32 + tn * 16 + r16;
      const float bv = bias[ng];
      #pragma unroll
      for (int r = 0; r < 8; ++r) {
        const int mg = m0 + wm * 32 + tm * 16 + r + hi * 8;
        z[((size_t)b * Ll + mg) * OUTD + ng] = acc[tm][tn][r] + bv;
      }
    }
  }
}

// ---------------------------------------------------------------------------
// Kernel 5: GLU + transpose back via LDS tile (coalesced on both sides):
//   out[b,h,l] = z[b,l,h] * sigmoid(z[b,l,h+H])
//   Tile: 32 h x 64 l per block. grid = (L/64, H/32, B)
// ---------------------------------------------------------------------------
__global__ __launch_bounds__(256)
void k_glu(const float* __restrict__ z, float* __restrict__ out) {
  __shared__ float ta[64][33];
  __shared__ float tg[64][33];
  const int l0 = blockIdx.x * 64;
  const int h0 = blockIdx.y * 32;
  const int b  = blockIdx.z;
  #pragma unroll
  for (int rep = 0; rep < 8; ++rep) {
    const int i  = threadIdx.x + rep * 256;   // 0..2047
    const int li = i >> 5;                    // 0..63
    const int hh = i & 31;                    // 0..31 (contiguous reads)
    const float* zr = z + ((size_t)b * Ll + l0 + li) * OUTD + h0;
    ta[li][hh] = zr[hh];
    tg[li][hh] = zr[hh + Hh];
  }
  __syncthreads();
  #pragma unroll
  for (int rep = 0; rep < 8; ++rep) {
    const int i  = threadIdx.x + rep * 256;
    const int hh = i >> 6;                    // 0..31
    const int li = i & 63;                    // 0..63 (contiguous writes)
    const float a = ta[li][hh];
    const float g = tg[li][hh];
    out[((size_t)b * Hh + h0 + hh) * Ll + l0 + li] = a * (1.0f / (1.0f + __expf(-g)));
  }
}

// ---------------------------------------------------------------------------
// Launch. Workspace layout (bytes):
//   kf  : C*H * 2*2049 f32            =  33,570,816
//   uf  : B*H * 2*2049 f32            = 134,283,264  (reused as z after conv)
//   act : B*CH*L bf16                 =  67,108,864
// total ~235 MB.
// ---------------------------------------------------------------------------
extern "C" void kernel_launch(void* const* d_in, const int* in_sizes, int n_in,
                              void* d_out, int out_size, void* d_ws, size_t ws_size,
                              hipStream_t stream) {
  const float* u    = (const float*)d_in[0];   // (B,H,L)
  const float* kern = (const float*)d_in[1];   // (C,H,2L)
  const float* Dp   = (const float*)d_in[2];   // (C,H)
  const float* W    = (const float*)d_in[3];   // (2H, C*H)
  const float* bias = (const float*)d_in[4];   // (2H,)
  float* out = (float*)d_out;                  // (B,H,L)

  char* ws = (char*)d_ws;
  const size_t kf_bytes = (size_t)Cc * Hh * 2 * NBIN * sizeof(float);
  const size_t uf_bytes = (size_t)Bb * Hh * 2 * NBIN * sizeof(float);
  float*          kf  = (float*)ws;
  float*          uf  = (float*)(ws + kf_bytes);
  unsigned short* act = (unsigned short*)(ws + kf_bytes + uf_bytes);
  float*          z   = uf;  // u_f dead after k_conv; z fits inside it

  k_prep_fft<<<Cc * Hh, NT, 0, stream>>>(kern, kf);
  k_ufft    <<<Bb * Hh, NT, 0, stream>>>(u, uf);
  k_conv    <<<Bb * Cc * Hh, NT, 0, stream>>>(uf, kf, u, Dp, act);
  k_gemm    <<<dim3(Ll / TM, OUTD / TN, Bb), 256, 0, stream>>>(act, W, bias, z);
  k_glu     <<<dim3(Ll / 64, Hh / 32, Bb), 256, 0, stream>>>(z, out);
}
